// GatFfn_76089640615986
// MI455X (gfx1250) — matrix operand order repaired and checked by reading, not compiled, not run on hardware
//
#include <hip/hip_runtime.h>

// ---------------------------------------------------------------------------
// GAT + MHA-proj + FFN for MI455X (gfx1250, wave32, WMMA bf16 16x16x32)
// ---------------------------------------------------------------------------

typedef __bf16 bf16;
typedef bf16  v8bf  __attribute__((ext_vector_type(8)));
typedef bf16  v16bf __attribute__((ext_vector_type(16)));
typedef float v8f   __attribute__((ext_vector_type(8)));

#define F_DIM 128
#define HEADS 4

// ---- helpers --------------------------------------------------------------

__device__ __forceinline__ unsigned short f2bfbits(float f) {
    unsigned int u = __float_as_uint(f);
    u += 0x7FFFu + ((u >> 16) & 1u);          // round-to-nearest-even
    return (unsigned short)(u >> 16);
}

// monotonic int mapping of IEEE fp32 (so atomicMax(int) == float max)
__device__ __forceinline__ int flipf(float f) {
    int b = __float_as_int(f);
    return b >= 0 ? b : (b ^ 0x7FFFFFFF);
}
__device__ __forceinline__ float unflipf(int b) {
    return __int_as_float(b >= 0 ? b : (b ^ 0x7FFFFFFF));
}

// two 16-byte chunks -> one 16-element bf16 fragment
__device__ __forceinline__ v16bf make_frag(const unsigned short* p0,
                                           const unsigned short* p1) {
    v8bf lo = *(const v8bf*)p0;
    v8bf hi = *(const v8bf*)p1;
    return __builtin_shufflevector(lo, hi, 0,1,2,3,4,5,6,7,8,9,10,11,12,13,14,15);
}

// ---- elementwise conversion ----------------------------------------------

__global__ void cvt_f32_to_bf16(const float* __restrict__ in,
                                unsigned short* __restrict__ out, long long n) {
    long long i = (long long)blockIdx.x * blockDim.x + threadIdx.x;
    long long stride = (long long)gridDim.x * blockDim.x;
    for (; i < n; i += stride) out[i] = f2bfbits(in[i]);
}

// ---- WMMA GEMM: C[M,NC] = A[M,K](bf16) * B[K,NC](bf16) + epilogue ---------
// block tile 128x128, 8 waves (4 M x 2 N), each wave 32x64 (2x4 WMMA tiles)

template <bool BIAS, bool RESID, bool LEAKY, bool SF32, bool SBF16>
__global__ __launch_bounds__(256)
void wmma_gemm_kernel(const unsigned short* __restrict__ A,
                      const unsigned short* __restrict__ B,
                      float* __restrict__ C,
                      unsigned short* __restrict__ Cbf,
                      const float* __restrict__ bias,
                      const float* __restrict__ resid,
                      int M, int K, int NC) {
    constexpr int KC   = 128;         // K staging chunk
    constexpr int KPAD = KC + 8;      // bank-stagger pad (16B granules)
    __shared__ unsigned short Bt[128 * KPAD];   // 34,816 B

    const int tid  = threadIdx.x;
    const int lane = tid & 31;
    const int wid  = tid >> 5;
    const int lmod = lane & 15;
    const int ldiv = lane >> 4;

    const int mBlock = blockIdx.x * 128;
    const int nBlock = blockIdx.y * 128;

    const int waveM = wid & 3;        // 0..3 -> 32-row slab
    const int waveN = wid >> 2;       // 0..1 -> 64-col slab
    const int mWave = mBlock + waveM * 32;
    const int nWaveL = waveN * 64;    // within block (LDS n index)

    v8f acc[2][4] = {};

    // clamp rows so EXEC stays all-ones through WMMA
    int rowA0 = mWave + lmod;        if (rowA0 >= M) rowA0 = M - 1;
    int rowA1 = mWave + 16 + lmod;   if (rowA1 >= M) rowA1 = M - 1;
    const unsigned short* Arow0 = A + (size_t)rowA0 * K + ldiv * 8;
    const unsigned short* Arow1 = A + (size_t)rowA1 * K + ldiv * 8;
    const unsigned short* BtL   = Bt + (nWaveL + lmod) * KPAD + ldiv * 8;

    for (int kc = 0; kc < K; kc += KC) {
        // stage B^T chunk: Bt[n][k] = B[kc+k][nBlock+n]
        for (int idx = tid; idx < KC * 128; idx += 256) {
            int k = idx >> 7, n = idx & 127;
            Bt[n * KPAD + k] = B[(size_t)(kc + k) * NC + nBlock + n];
        }
        __syncthreads();

        #pragma unroll
        for (int k0 = 0; k0 < KC; k0 += 32) {
            const int kg = kc + k0;
            // A frag: lane holds K = {kg+ldiv*8 .. +8} and {+16 .. +8}
            v16bf a0 = make_frag(Arow0 + kg, Arow0 + kg + 16);
            v16bf a1 = make_frag(Arow1 + kg, Arow1 + kg + 16);
            v16bf bfr[4];
            #pragma unroll
            for (int nt = 0; nt < 4; nt++) {
                // B frag: lane holds K = {ldiv*8 .. +8} and {+16 .. +8}
                const unsigned short* pb = BtL + nt * 16 * KPAD + k0;
                bfr[nt] = make_frag(pb, pb + 16);
            }
            #pragma unroll
            for (int nt = 0; nt < 4; nt++) {
                acc[0][nt] = __builtin_amdgcn_wmma_f32_16x16x32_bf16(
                    false, a0, false, bfr[nt], (short)0, acc[0][nt], false, false);
                acc[1][nt] = __builtin_amdgcn_wmma_f32_16x16x32_bf16(
                    false, a1, false, bfr[nt], (short)0, acc[1][nt], false, false);
            }
        }
        __syncthreads();
    }

    // epilogue: (+bias) (leaky 0.01) (+resid) -> f32 and/or bf16
    #pragma unroll
    for (int mt = 0; mt < 2; mt++) {
        #pragma unroll
        for (int nt = 0; nt < 4; nt++) {
            const int nG = nBlock + nWaveL + nt * 16 + lmod;
            float bv = 0.0f;
            if (BIAS) bv = bias[nG];
            #pragma unroll
            for (int r = 0; r < 8; r++) {
                const int mG = mWave + mt * 16 + ldiv * 8 + r;
                if (mG < M) {
                    float v = acc[mt][nt][r];
                    if (BIAS)  v += bv;
                    if (LEAKY) v = v > 0.0f ? v : 0.01f * v;
                    if (RESID) v += resid[(size_t)mG * NC + nG];
                    if (SF32)  C[(size_t)mG * NC + nG] = v;
                    if (SBF16) Cbf[(size_t)mG * NC + nG] = f2bfbits(v);
                }
            }
        }
    }
}

// ---- attention scores: a_src/a_dst[n,h] = dot(xp[n,h,:], att[h,:]) --------

__global__ void scores_kernel(const float* __restrict__ xp,
                              const float* __restrict__ att_src,
                              const float* __restrict__ att_dst,
                              float* __restrict__ a_src,
                              float* __restrict__ a_dst, int Nn) {
    __shared__ float s1[128], s2[128];
    const int n = blockIdx.x, t = threadIdx.x;
    for (int h = 0; h < HEADS; h++) {
        float xv = xp[(size_t)n * 512 + h * 128 + t];
        s1[t] = xv * att_src[h * 128 + t];
        s2[t] = xv * att_dst[h * 128 + t];
        __syncthreads();
        for (int off = 64; off > 0; off >>= 1) {
            if (t < off) { s1[t] += s1[t + off]; s2[t] += s2[t + off]; }
            __syncthreads();
        }
        if (t == 0) { a_src[n * 4 + h] = s1[0]; a_dst[n * 4 + h] = s2[0]; }
        __syncthreads();
    }
}

// ---- init: m = flip(-inf), denom = 0, agg = 0 -----------------------------

__global__ void attn_init_kernel(int* __restrict__ m_i, float* __restrict__ denom,
                                 float* __restrict__ agg, long long nAgg, long long nS) {
    long long i = (long long)blockIdx.x * blockDim.x + threadIdx.x;
    long long stride = (long long)gridDim.x * blockDim.x;
    const int ninf = flipf(-__builtin_inff());
    for (; i < nAgg; i += stride) {
        agg[i] = 0.0f;
        if (i < nS) { m_i[i] = ninf; denom[i] = 0.0f; }
    }
}

// ---- edge pass A: segment max (int-flipped fp32 atomicMax) ----------------

__global__ void edge_max_kernel(const int* __restrict__ src, const int* __restrict__ dst,
                                const float* __restrict__ a_src, const float* __restrict__ a_dst,
                                int* __restrict__ m_i, long long Ee, long long EE) {
    long long e = (long long)blockIdx.x * blockDim.x + threadIdx.x;
    if (e >= EE) return;
    int s, d;
    if (e < Ee) { s = src[e]; d = dst[e]; } else { s = d = (int)(e - Ee); }
    #pragma unroll
    for (int h = 0; h < HEADS; h++) {
        float al = a_src[s * 4 + h] + a_dst[d * 4 + h];
        al = al > 0.0f ? al : 0.2f * al;
        atomicMax(&m_i[d * 4 + h], flipf(al));
    }
}

// ---- edge pass B: denom = segment_sum(exp(alpha - m)) ---------------------

__global__ void edge_sum_kernel(const int* __restrict__ src, const int* __restrict__ dst,
                                const float* __restrict__ a_src, const float* __restrict__ a_dst,
                                const int* __restrict__ m_i, float* __restrict__ denom,
                                long long Ee, long long EE) {
    long long e = (long long)blockIdx.x * blockDim.x + threadIdx.x;
    if (e >= EE) return;
    int s, d;
    if (e < Ee) { s = src[e]; d = dst[e]; } else { s = d = (int)(e - Ee); }
    #pragma unroll
    for (int h = 0; h < HEADS; h++) {
        float al = a_src[s * 4 + h] + a_dst[d * 4 + h];
        al = al > 0.0f ? al : 0.2f * al;
        atomicAdd(&denom[d * 4 + h], __expf(al - unflipf(m_i[d * 4 + h])));
    }
}

// ---- edge pass C: agg[dst] += xp[src] * att  (one wave per edge) ----------
// xp (102 MB) + agg (102 MB) ~ L2-resident (192 MB) -> fp32 atomics at L2.

__global__ void edge_scatter_kernel(const int* __restrict__ src, const int* __restrict__ dst,
                                    const float* __restrict__ a_src, const float* __restrict__ a_dst,
                                    const int* __restrict__ m_i, const float* __restrict__ denom,
                                    const float* __restrict__ xp, float* __restrict__ agg,
                                    long long Ee, long long EE) {
    long long gid = (long long)blockIdx.x * 8 + (threadIdx.x >> 5);
    if (gid >= EE) return;
    const int lane = threadIdx.x & 31;
    int s, d;
    if (gid < Ee) { s = src[gid]; d = dst[gid]; } else { s = d = (int)(gid - Ee); }
    #pragma unroll
    for (int h = 0; h < HEADS; h++) {
        float al = a_src[s * 4 + h] + a_dst[d * 4 + h];
        al = al > 0.0f ? al : 0.2f * al;
        float att = __expf(al - unflipf(m_i[d * 4 + h])) / denom[d * 4 + h];
        const float4 v = *(const float4*)(xp + (size_t)s * 512 + h * 128 + lane * 4);
        float* ag = agg + (size_t)d * 512 + h * 128 + lane * 4;
        atomicAdd(ag + 0, v.x * att);
        atomicAdd(ag + 1, v.y * att);
        atomicAdd(ag + 2, v.z * att);
        atomicAdd(ag + 3, v.w * att);
    }
}

// ---- GAT epilogue: leaky(agg + bias_gat) -> bf16 --------------------------

__global__ void gat_bias_act_kernel(const float* __restrict__ agg,
                                    const float* __restrict__ bias,
                                    unsigned short* __restrict__ out_bf, long long total) {
    long long i = (long long)blockIdx.x * blockDim.x + threadIdx.x;
    long long stride = (long long)gridDim.x * blockDim.x;
    for (; i < total; i += stride) {
        float v = agg[i] + bias[i & 511];
        v = v > 0.0f ? v : 0.01f * v;
        out_bf[i] = f2bfbits(v);
    }
}

// ---------------------------------------------------------------------------

extern "C" void kernel_launch(void* const* d_in, const int* in_sizes, int n_in,
                              void* d_out, int out_size, void* d_ws, size_t ws_size,
                              hipStream_t stream) {
    const float* x        = (const float*)d_in[0];
    const int*   ei       = (const int*)d_in[1];
    const float* W_gat    = (const float*)d_in[2];
    const float* att_src  = (const float*)d_in[3];
    const float* att_dst  = (const float*)d_in[4];
    const float* bias_gat = (const float*)d_in[5];
    const float* W_mha    = (const float*)d_in[6];
    const float* b_mha    = (const float*)d_in[7];
    const float* W1       = (const float*)d_in[8];
    const float* b1       = (const float*)d_in[9];
    const float* W2       = (const float*)d_in[10];
    const float* b2       = (const float*)d_in[11];
    float* out = (float*)d_out;

    const size_t Nn = (size_t)in_sizes[0] / F_DIM;
    const size_t Ee = (size_t)in_sizes[1] / 2;
    const long long EE = (long long)(Ee + Nn);
    const int* srcI = ei;
    const int* dstI = ei + Ee;

    // workspace layout (aliased phases)
    char* w = (char*)d_ws;
    const size_t szXp = Nn * 512 * sizeof(float);          // 102.4 MB
    float* xp = (float*)w;                                  // phase 1-2
    unsigned short* agg_bf = (unsigned short*)w;            // phase 2b (xp dead)
    char* r2 = w + szXp;
    unsigned short* x_bf = (unsigned short*)r2;             // phase 1 only
    float* agg  = (float*)r2;                               // phase 2 (x_bf dead)
    float* hbuf = (float*)r2;                               // phase 3 (agg dead)
    unsigned short* h_bf   = (unsigned short*)(r2 + Nn * 128 * 4);
    unsigned short* mid_bf = (unsigned short*)(r2 + Nn * 128 * 4 + Nn * 128 * 2);
    char* r3 = w + 2 * szXp;
    float* a_srcv = (float*)r3;
    float* a_dstv = a_srcv + Nn * 4;
    int*   m_i    = (int*)(a_dstv + Nn * 4);
    float* denomv = (float*)(m_i + Nn * 4);
    unsigned short* wgat_bf = (unsigned short*)(denomv + Nn * 4);
    unsigned short* wmha_bf = wgat_bf + 128 * 512;
    unsigned short* w1_bf   = wmha_bf + 512 * 128;
    unsigned short* w2_bf   = w1_bf + 128 * 128;

    // 1) bf16 conversions
    cvt_f32_to_bf16<<<(unsigned)((Nn * 128 + 255) / 256), 256, 0, stream>>>(x, x_bf, (long long)(Nn * 128));
    cvt_f32_to_bf16<<<256, 256, 0, stream>>>(W_gat, wgat_bf, 128 * 512);
    cvt_f32_to_bf16<<<256, 256, 0, stream>>>(W_mha, wmha_bf, 512 * 128);
    cvt_f32_to_bf16<<<64, 256, 0, stream>>>(W1, w1_bf, 128 * 128);
    cvt_f32_to_bf16<<<64, 256, 0, stream>>>(W2, w2_bf, 128 * 128);

    const unsigned mBlocks = (unsigned)((Nn + 127) / 128);

    // 2) xp = x @ W_gat  (raw f32 out)
    wmma_gemm_kernel<false, false, false, true, false>
        <<<dim3(mBlocks, 4), 256, 0, stream>>>(x_bf, wgat_bf, xp, nullptr,
                                               nullptr, nullptr, (int)Nn, 128, 512);

    // 3) attention scores
    scores_kernel<<<(unsigned)Nn, 128, 0, stream>>>(xp, att_src, att_dst, a_srcv, a_dstv, (int)Nn);

    // 4) init softmax stats + aggregation buffer
    const long long nAgg = (long long)Nn * 512;
    attn_init_kernel<<<(unsigned)((nAgg + 255) / 256), 256, 0, stream>>>(m_i, denomv, agg, nAgg, (long long)(Nn * 4));

    // 5-7) segment softmax + weighted scatter (includes self-loops, idx >= E)
    const unsigned gE = (unsigned)((EE + 255) / 256);
    edge_max_kernel<<<gE, 256, 0, stream>>>(srcI, dstI, a_srcv, a_dstv, m_i, (long long)Ee, EE);
    edge_sum_kernel<<<gE, 256, 0, stream>>>(srcI, dstI, a_srcv, a_dstv, m_i, denomv, (long long)Ee, EE);
    edge_scatter_kernel<<<(unsigned)((EE + 7) / 8), 256, 0, stream>>>(srcI, dstI, a_srcv, a_dstv,
                                                                      m_i, denomv, xp, agg,
                                                                      (long long)Ee, EE);

    // 8) leaky(agg + bias_gat) -> bf16 (into dead xp region)
    gat_bias_act_kernel<<<(unsigned)((nAgg + 255) / 256), 256, 0, stream>>>(agg, bias_gat, agg_bf, nAgg);

    // 9) h = agg_bf @ W_mha + b_mha + x   (f32 + bf16 outputs)
    wmma_gemm_kernel<true, true, false, true, true>
        <<<dim3(mBlocks, 1), 256, 0, stream>>>(agg_bf, wmha_bf, hbuf, h_bf,
                                               b_mha, x, (int)Nn, 512, 128);

    // 10) mid = leaky(h @ W1 + b1)  (bf16 only)
    wmma_gemm_kernel<true, false, true, false, true>
        <<<dim3(mBlocks, 1), 256, 0, stream>>>(h_bf, w1_bf, nullptr, mid_bf,
                                               b1, nullptr, (int)Nn, 128, 128);

    // 11) out = mid @ W2 + b2 + h
    wmma_gemm_kernel<true, true, false, true, false>
        <<<dim3(mBlocks, 1), 256, 0, stream>>>(mid_bf, w2_bf, out, nullptr,
                                               b2, hbuf, (int)Nn, 128, 128);
}
